// TransformerDSACaptioner_24584392802494
// MI455X (gfx1250) — compile-verified, matrix operand order checked by reading, not run on hardware
//
#include <hip/hip_runtime.h>
#include <hip/hip_bf16.h>
#include <math.h>

// ---------------------------------------------------------------------------
// Model dims (fixed by the reference)
// ---------------------------------------------------------------------------
#define BB 4
#define QQ 16
#define LCC 32
#define DD 512
#define MM 8
#define LVV 4
#define PP 4
#define VV 30000
#define DFFD 1024
#define NLL 2
#define DHH 64
#define LINN 1920              // 1024+512+256+128
#define NTOK 2048              // B*Q*LC
#define NVOC 30001             // V+1

#define GWAVES 4               // waves per GEMM block (256 rows of M per block)

typedef __attribute__((ext_vector_type(16))) __bf16 v16bf;
typedef __attribute__((ext_vector_type(8)))  float  v8f;

// ---------------------------------------------------------------------------
// WMMA GEMM:  C[M x N] = A[M x K] @ W[K x N] + bias, opt relu/rowmask.
// Block = 4 waves, each wave computes a 64x16 slab (4 M-tiles sharing one B
// operand -> 4x v_wmma_f32_16x16x32_bf16 per K-step per wave).
// The 32x16 B tile of W is cooperatively loaded ONCE per block (coalesced
// along N), converted to bf16 and stored transposed [n][k] in LDS; each lane
// then grabs its B fragment with two 16-byte DS reads. Double-buffered, one
// barrier per K-step.
// Operand lane layouts per ISA 7.12.2:
//   A (16-bit, 16x32): lane l(0..15): row=l, K(e)= e | 16+(e-8);
//                      lane l(16..31): row=l-16, K(e)= 8+e | 24+(e-8)
//   B: same layout applied to W columns.  D: lane l -> col l&15,
//      VGPR r -> row r + 8*(l>=16).
// Requires M % 256 == 0 (true for all call sites: 2048, 7680).
// ---------------------------------------------------------------------------
__global__ void wmma_gemm_kernel(const float* __restrict__ A, int lda,
                                 const float* __restrict__ W, int ldw,
                                 const float* __restrict__ bias,
                                 float* __restrict__ C, int ldc,
                                 int Ndim, int Kdim, int relu,
                                 const unsigned char* __restrict__ rowmask)
{
    const int tid  = threadIdx.x;          // 0..127
    const int wave = tid >> 5;
    const int lane = tid & 31;
    const int half = lane >> 4;            // 0 or 1
    const int l15  = lane & 15;
    const int tn   = blockIdx.y * 16;
    const int tm   = blockIdx.x * (GWAVES * 64) + wave * 64;
    const int bcol = tn + l15;
    const bool bvalid = (bcol < Ndim);
    const int  kA  = half * 8;

    // B tile staged transposed: bsh[buf][n][k], row stride 32 bf16 = 64B
    __shared__ __align__(16) __bf16 bsh[2][16][32];

    v8f acc[4] = {{}, {}, {}, {}};

    int buf = 0;
    for (int k0 = 0; k0 < Kdim; k0 += 32, buf ^= 1) {
        // ---- cooperative, coalesced fill of the 32x16 B tile (bf16, [n][k])
        for (int idx = tid; idx < 512; idx += 128) {
            const int k = idx >> 4, n = idx & 15;
            const int col = tn + n;
            const int cc  = (col < Ndim) ? col : (Ndim - 1);   // clamp (discarded)
            bsh[buf][n][k] = (__bf16)W[(size_t)(k0 + k) * ldw + cc];
            if (k0 + 32 < Kdim)   // gfx1250 global_prefetch of next K tile
                __builtin_prefetch(&W[(size_t)(k0 + 32 + k) * ldw + cc], 0, 1);
        }
        __syncthreads();

        // ---- per-lane B fragment: two 16B DS reads (K runs are contiguous)
        union { uint4 u[2]; v16bf v; } bu;
        bu.u[0] = *(const uint4*)&bsh[buf][l15][kA];
        bu.u[1] = *(const uint4*)&bsh[buf][l15][kA + 16];
        const v16bf b = bu.v;

        // ---- 4 A operands (rows, contiguous float4 loads) + 4 WMMAs
#pragma unroll
        for (int mt = 0; mt < 4; ++mt) {
            const float* Ap = A + (size_t)(tm + mt * 16 + l15) * lda + k0 + kA;
            const float4 x0 = *(const float4*)(Ap);
            const float4 x1 = *(const float4*)(Ap + 4);
            const float4 x2 = *(const float4*)(Ap + 16);
            const float4 x3 = *(const float4*)(Ap + 20);
            v16bf a;
            a[0]  = (__bf16)x0.x; a[1]  = (__bf16)x0.y;
            a[2]  = (__bf16)x0.z; a[3]  = (__bf16)x0.w;
            a[4]  = (__bf16)x1.x; a[5]  = (__bf16)x1.y;
            a[6]  = (__bf16)x1.z; a[7]  = (__bf16)x1.w;
            a[8]  = (__bf16)x2.x; a[9]  = (__bf16)x2.y;
            a[10] = (__bf16)x2.z; a[11] = (__bf16)x2.w;
            a[12] = (__bf16)x3.x; a[13] = (__bf16)x3.y;
            a[14] = (__bf16)x3.z; a[15] = (__bf16)x3.w;
            // D = A x B + C (8 args: neg_a, A, neg_b, B, c_mod, C, reuse_a, reuse_b)
            acc[mt] = __builtin_amdgcn_wmma_f32_16x16x32_bf16(
                false, a, false, b, (short)0, acc[mt], false, false);
        }
    }

    if (bvalid) {
        const float bb = bias ? bias[bcol] : 0.0f;
#pragma unroll
        for (int mt = 0; mt < 4; ++mt) {
#pragma unroll
            for (int r = 0; r < 8; ++r) {
                const int row = tm + mt * 16 + half * 8 + r;
                float v = acc[mt][r] + bb;
                if (relu) v = fmaxf(v, 0.0f);
                if (rowmask && rowmask[row]) v = 0.0f;
                C[(size_t)row * ldc + bcol] = v;
            }
        }
    }
}

// ---------------------------------------------------------------------------
// Embedding + sinusoidal positional encoding: hs[t,d] = embed_w[seq[t],d]+PE
// ---------------------------------------------------------------------------
__global__ void embed_kernel(const int* __restrict__ seq,
                             const float* __restrict__ embed_w,
                             float* __restrict__ hs)
{
    const int t   = blockIdx.x;           // 0..2047
    const int lc  = t & 31;
    const int tok = seq[t];
    const float kln = -9.21034037197618f / (float)DD;   // -ln(10000)/D
    for (int d = threadIdx.x; d < DD; d += blockDim.x) {
        const int de = d & ~1;
        const float ang = (float)lc * expf((float)de * kln);
        const float pe  = (d & 1) ? cosf(ang) : sinf(ang);
        hs[(size_t)t * DD + d] = embed_w[(size_t)tok * DD + d] + pe;
    }
}

// ---------------------------------------------------------------------------
// Causal dense self-attention per (sequence n, head h). LC=32, DH=64.
// ---------------------------------------------------------------------------
__global__ void attn_kernel(const float* __restrict__ q,
                            const float* __restrict__ k,
                            const float* __restrict__ v,
                            float* __restrict__ ctx)
{
    const int n = blockIdx.x >> 3;
    const int h = blockIdx.x & 7;
    __shared__ float qs[LCC][DHH], ks[LCC][DHH], vs[LCC][DHH];
    __shared__ float sc[LCC][LCC + 1];
    const int tid  = threadIdx.x;          // 128 threads
    const int base = n * LCC;
    const int col0 = h * DHH;

    for (int idx = tid; idx < LCC * DHH; idx += 128) {
        const int i = idx >> 6, d = idx & 63;
        const size_t off = (size_t)(base + i) * DD + col0 + d;
        qs[i][d] = q[off]; ks[i][d] = k[off]; vs[i][d] = v[off];
    }
    __syncthreads();

    for (int idx = tid; idx < LCC * LCC; idx += 128) {
        const int i = idx >> 5, j = idx & 31;
        float s = 0.f;
#pragma unroll 8
        for (int d = 0; d < DHH; ++d) s += qs[i][d] * ks[j][d];
        s *= 0.125f;                           // 1/sqrt(64)
        if (j > i) s += -10000.0f;             // causal additive bias
        sc[i][j] = s;
    }
    __syncthreads();

    if (tid < LCC) {                           // softmax row tid
        float mx = -3.4e38f;
        for (int j = 0; j < LCC; ++j) mx = fmaxf(mx, sc[tid][j]);
        float sum = 0.f;
        for (int j = 0; j < LCC; ++j) { const float e = expf(sc[tid][j] - mx); sc[tid][j] = e; sum += e; }
        const float inv = 1.0f / sum;
        for (int j = 0; j < LCC; ++j) sc[tid][j] *= inv;
    }
    __syncthreads();

    for (int idx = tid; idx < LCC * DHH; idx += 128) {
        const int i = idx >> 6, d = idx & 63;
        float s = 0.f;
#pragma unroll 8
        for (int j = 0; j < LCC; ++j) s += sc[i][j] * vs[j][d];
        ctx[(size_t)(base + i) * DD + col0 + d] = s;
    }
}

// ---------------------------------------------------------------------------
// LayerNorm over D=512 with optional residual: out = LN(x (+res)) * g + b
// (out may alias x; the row is cached in LDS before writing)
// ---------------------------------------------------------------------------
__global__ void ln_kernel(const float* __restrict__ x,
                          const float* __restrict__ res,
                          const float* __restrict__ g,
                          const float* __restrict__ b,
                          float* __restrict__ out)
{
    const int t = blockIdx.x;
    __shared__ float red[256];
    __shared__ float row[DD];
    const int tid = threadIdx.x;

    float s = 0.f;
    for (int d = tid; d < DD; d += 256) {
        float vv = x[(size_t)t * DD + d];
        if (res) vv += res[(size_t)t * DD + d];
        row[d] = vv; s += vv;
    }
    red[tid] = s; __syncthreads();
    for (int o = 128; o > 0; o >>= 1) { if (tid < o) red[tid] += red[tid + o]; __syncthreads(); }
    const float mean = red[0] / (float)DD;
    __syncthreads();

    float vsum = 0.f;
    for (int d = tid; d < DD; d += 256) { const float dv = row[d] - mean; vsum += dv * dv; }
    red[tid] = vsum; __syncthreads();
    for (int o = 128; o > 0; o >>= 1) { if (tid < o) red[tid] += red[tid + o]; __syncthreads(); }
    const float rstd = rsqrtf(red[0] / (float)DD + 1e-5f);

    for (int d = tid; d < DD; d += 256)
        out[(size_t)t * DD + d] = (row[d] - mean) * rstd * g[d] + b[d];
}

// ---------------------------------------------------------------------------
// cat = [hs(t), query(t/32)]  (2048 x 1024)
// ---------------------------------------------------------------------------
__global__ void concat_kernel(const float* __restrict__ hs,
                              const float* __restrict__ query,
                              float* __restrict__ cat)
{
    const int t = blockIdx.x;
    const int n = t >> 5;                       // B*Q index
    for (int c = threadIdx.x; c < 2 * DD; c += 256) {
        cat[(size_t)t * (2 * DD) + c] =
            (c < DD) ? hs[(size_t)t * DD + c] : query[(size_t)n * DD + (c - DD)];
    }
}

// ---------------------------------------------------------------------------
// Softmax over the last 16 (LV*P) attention weights per (token, head)
// ---------------------------------------------------------------------------
__global__ void awsoftmax_kernel(float* __restrict__ aw)
{
    const int item = blockIdx.x * blockDim.x + threadIdx.x;
    if (item >= NTOK * MM) return;
    const int t = item >> 3, m = item & 7;
    float* p = aw + (size_t)t * (MM * LVV * PP) + m * (LVV * PP);
    float mx = -3.4e38f;
#pragma unroll
    for (int j = 0; j < LVV * PP; ++j) mx = fmaxf(mx, p[j]);
    float s = 0.f;
#pragma unroll
    for (int j = 0; j < LVV * PP; ++j) { const float e = expf(p[j] - mx); p[j] = e; s += e; }
    const float inv = 1.0f / s;
#pragma unroll
    for (int j = 0; j < LVV * PP; ++j) p[j] *= inv;
}

// ---------------------------------------------------------------------------
// 1-D multiscale deformable sampling (linear interp, zero padding, clipped
// gather) — exact reference semantics.  grid=(2048, 8), block=64 (dh).
// ---------------------------------------------------------------------------
__global__ void deform_kernel(const float* __restrict__ val,   // (B,LIN,D)
                              const float* __restrict__ off,   // (2048,128)
                              const float* __restrict__ aw,    // (2048,128)
                              const float* __restrict__ refr,  // (B,Q,1)
                              const float* __restrict__ vr,    // (B,LV)
                              float* __restrict__ out)         // (2048,512)
{
    const int t  = blockIdx.x;
    const int m  = blockIdx.y;
    const int dh = threadIdx.x;
    const int b  = t >> 9;                 // / (Q*LC)
    const int qq = (t & 511) >> 5;         // / LC
    const float refq = refr[b * QQ + qq];

    const int Ts[4]     = {1024, 512, 256, 128};
    const int starts[4] = {0, 1024, 1536, 1792};

    const float* vbase = val + (size_t)b * LINN * DD + m * DHH + dh;
    float acc = 0.f;
#pragma unroll
    for (int l = 0; l < 4; ++l) {
        const float r  = refq * vr[b * LVV + l];
        const int   T  = Ts[l];
        const float Tf = (float)T;
#pragma unroll
        for (int p = 0; p < 4; ++p) {
            const int   fi  = m * 16 + l * 4 + p;
            const float o   = off[(size_t)t * 128 + fi];
            const float a   = aw [(size_t)t * 128 + fi];
            const float pos = (r + o / Tf) * Tf - 0.5f;
            const float fl  = floorf(pos);
            const float tt  = pos - fl;
            const int i0 = (int)fl, i1 = i0 + 1;
            const float w0 = (1.0f - tt) * ((i0 >= 0 && i0 < T) ? 1.0f : 0.0f);
            const float w1 = tt          * ((i1 >= 0 && i1 < T) ? 1.0f : 0.0f);
            const int c0 = min(max(i0, 0), T - 1);
            const int c1 = min(max(i1, 0), T - 1);
            const float s = w0 * vbase[(size_t)(starts[l] + c0) * DD]
                          + w1 * vbase[(size_t)(starts[l] + c1) * DD];
            acc += a * s;
        }
    }
    out[(size_t)t * DD + m * DHH + dh] = acc;
}

// ---------------------------------------------------------------------------
// In-place log_softmax over rows of length NVOC
// ---------------------------------------------------------------------------
__global__ void logsoftmax_kernel(float* __restrict__ out)
{
    const int t = blockIdx.x;
    float* row = out + (size_t)t * NVOC;
    __shared__ float red[256];
    const int tid = threadIdx.x;

    float mx = -3.4e38f;
    for (int j = tid; j < NVOC; j += 256) mx = fmaxf(mx, row[j]);
    red[tid] = mx; __syncthreads();
    for (int o = 128; o > 0; o >>= 1) { if (tid < o) red[tid] = fmaxf(red[tid], red[tid + o]); __syncthreads(); }
    mx = red[0]; __syncthreads();

    float s = 0.f;
    for (int j = tid; j < NVOC; j += 256) s += expf(row[j] - mx);
    red[tid] = s; __syncthreads();
    for (int o = 128; o > 0; o >>= 1) { if (tid < o) red[tid] += red[tid + o]; __syncthreads(); }
    const float lse = mx + logf(red[0]);

    for (int j = tid; j < NVOC; j += 256) row[j] -= lse;
}

// ---------------------------------------------------------------------------
// Host-side orchestration
// ---------------------------------------------------------------------------
static inline void gemm(const float* A, int lda, const float* W, int ldw,
                        const float* bias, float* C, int ldc,
                        int Mdim, int Ndim, int Kdim, int relu,
                        const unsigned char* rowmask, hipStream_t stream)
{
    dim3 grid(Mdim / (GWAVES * 64), (Ndim + 15) / 16);  // block = 256 x 16 slab
    wmma_gemm_kernel<<<grid, GWAVES * 32, 0, stream>>>(A, lda, W, ldw, bias, C,
                                                       ldc, Ndim, Kdim, relu,
                                                       rowmask);
}

extern "C" void kernel_launch(void* const* d_in, const int* in_sizes, int n_in,
                              void* d_out, int out_size, void* d_ws, size_t ws_size,
                              hipStream_t stream)
{
    (void)in_sizes; (void)n_in; (void)out_size; (void)ws_size;

    const float* query  = (const float*)d_in[0];
    const float* refr   = (const float*)d_in[1];
    const float* vr     = (const float*)d_in[2];
    const float* memory = (const float*)d_in[3];
    const unsigned char* mask = (const unsigned char*)d_in[6];
    const int*   seq    = (const int*)d_in[7];

    const float* Wq     = (const float*)d_in[8];
    const float* bq     = (const float*)d_in[9];
    const float* Wk     = (const float*)d_in[10];
    const float* bk     = (const float*)d_in[11];
    const float* Wv     = (const float*)d_in[12];
    const float* bv     = (const float*)d_in[13];
    const float* ln1_g  = (const float*)d_in[14];
    const float* ln1_b  = (const float*)d_in[15];
    const float* proj_w = (const float*)d_in[16];
    const float* proj_b = (const float*)d_in[17];
    const float* off_w  = (const float*)d_in[18];
    const float* off_b  = (const float*)d_in[19];
    const float* aw_w   = (const float*)d_in[20];
    const float* aw_b   = (const float*)d_in[21];
    const float* val_w  = (const float*)d_in[22];
    const float* val_b  = (const float*)d_in[23];
    const float* out_w  = (const float*)d_in[24];
    const float* out_b  = (const float*)d_in[25];
    const float* ln2_g  = (const float*)d_in[26];
    const float* ln2_b  = (const float*)d_in[27];
    const float* ffn1_w = (const float*)d_in[28];
    const float* ffn1_b = (const float*)d_in[29];
    const float* ffn2_w = (const float*)d_in[30];
    const float* ffn2_b = (const float*)d_in[31];
    const float* ln3_g  = (const float*)d_in[32];
    const float* ln3_b  = (const float*)d_in[33];
    const float* embed_w  = (const float*)d_in[34];
    const float* logits_w = (const float*)d_in[35];
    const float* logits_b = (const float*)d_in[36];

    // ---- workspace layout (floats) ----
    float* ws = (float*)d_ws;
    float* hs   = ws;                                   // 2048*512
    float* bufA = hs   + (size_t)NTOK * DD;             // 2048*1024 (cat / ffn hid)
    float* bufB = bufA + (size_t)NTOK * 2 * DD;         // q
    float* bufC = bufB + (size_t)NTOK * DD;             // k
    float* bufD = bufC + (size_t)NTOK * DD;             // v
    float* bufE = bufD + (size_t)NTOK * DD;             // ctx / deform out / ffn out
    float* bufF = bufE + (size_t)NTOK * DD;             // hs2 (post-proj)
    float* bufG = bufF + (size_t)NTOK * DD;             // cross
    float* val  = bufG + (size_t)NTOK * DD;             // 7680*512
    float* offb = val  + (size_t)BB * LINN * DD;        // 2048*128
    float* awb  = offb + (size_t)NTOK * 128;            // 2048*128

    // ---- embedding + positional encoding ----
    embed_kernel<<<NTOK, 256, 0, stream>>>(seq, embed_w, hs);

    for (int i = 0; i < NLL; ++i) {
        const size_t wDD   = (size_t)i * DD * DD;
        const size_t bD    = (size_t)i * DD;
        // ---- self-attention qkv ----
        gemm(hs, DD, Wq + wDD, DD, bq + bD, bufB, DD, NTOK, DD, DD, 0, nullptr, stream);
        gemm(hs, DD, Wk + wDD, DD, bk + bD, bufC, DD, NTOK, DD, DD, 0, nullptr, stream);
        gemm(hs, DD, Wv + wDD, DD, bv + bD, bufD, DD, NTOK, DD, DD, 0, nullptr, stream);
        attn_kernel<<<(BB * QQ) * MM, 128, 0, stream>>>(bufB, bufC, bufD, bufE);
        ln_kernel<<<NTOK, 256, 0, stream>>>(hs, bufE, ln1_g + bD, ln1_b + bD, hs);

        // ---- concat event query + project 2D -> D ----
        concat_kernel<<<NTOK, 256, 0, stream>>>(hs, query, bufA);
        gemm(bufA, 2 * DD, proj_w + (size_t)i * 2 * DD * DD, DD, proj_b + bD,
             bufF, DD, NTOK, DD, 2 * DD, 0, nullptr, stream);

        // ---- deformable cross-attention ----
        gemm(memory, DD, val_w + wDD, DD, val_b + bD, val, DD,
             BB * LINN, DD, DD, 0, mask, stream);
        gemm(bufF, DD, off_w + (size_t)i * DD * 128, 128, off_b + (size_t)i * 128,
             offb, 128, NTOK, 128, DD, 0, nullptr, stream);
        gemm(bufF, DD, aw_w + (size_t)i * DD * 128, 128, aw_b + (size_t)i * 128,
             awb, 128, NTOK, 128, DD, 0, nullptr, stream);
        awsoftmax_kernel<<<(NTOK * MM + 255) / 256, 256, 0, stream>>>(awb);
        deform_kernel<<<dim3(NTOK, MM), DHH, 0, stream>>>(val, offb, awb, refr, vr, bufE);
        gemm(bufE, DD, out_w + wDD, DD, out_b + bD, bufG, DD, NTOK, DD, DD, 0, nullptr, stream);
        ln_kernel<<<NTOK, 256, 0, stream>>>(bufF, bufG, ln2_g + bD, ln2_b + bD, hs);

        // ---- FFN ----
        gemm(hs, DD, ffn1_w + (size_t)i * DD * DFFD, DFFD, ffn1_b + (size_t)i * DFFD,
             bufA, DFFD, NTOK, DFFD, DD, 1, nullptr, stream);
        gemm(bufA, DFFD, ffn2_w + (size_t)i * DFFD * DD, DD, ffn2_b + bD,
             bufE, DD, NTOK, DD, DFFD, 0, nullptr, stream);
        ln_kernel<<<NTOK, 256, 0, stream>>>(hs, bufE, ln3_g + bD, ln3_b + bD, hs);
    }

    // ---- vocab projection + log_softmax (in-place on d_out) ----
    float* logits = (float*)d_out;
    gemm(hs, DD, logits_w, NVOC, logits_b, logits, NVOC, NTOK, NVOC, DD, 0, nullptr, stream);
    logsoftmax_kernel<<<NTOK, 256, 0, stream>>>(logits);
}
